// HierarchicalMemory_50311246905662
// MI455X (gfx1250) — matrix operand order repaired and checked by reading, not compiled, not run on hardware
//
#include <hip/hip_runtime.h>
#include <hip/hip_fp16.h>

// ---------------------------------------------------------------------------
// HierarchicalMemory on MI455X (gfx1250): f16 WMMA pipeline, software-pipelined
//   B=8, S=1024, H=1024, M=16384   (8192 query rows)
// ---------------------------------------------------------------------------

typedef __attribute__((ext_vector_type(16))) _Float16 v16h;
typedef __attribute__((ext_vector_type(8)))  float    v8f;

#define NEG_INF   (-1.0e9f)
#define INV_TEMP  (10.0f)
#define HDIM      1024
#define MSLOTS    16384
#define NROWS     8192

static __device__ __forceinline__ v8f wmma_f16(v16h a, v16h b, v8f c) {
  // v_wmma_f32_16x16x32_f16: (neg_a, A, neg_b, B, c_mod, C, reuse_a, reuse_b)
  return __builtin_amdgcn_wmma_f32_16x16x32_f16(false, a, false, b, (short)0, c,
                                                false, false);
}

// 16-bit A-matrix 16x32 layout (matches B): lane L<16 holds K=k..k+15,
// lane L>=16 holds K=k+16..k+31. Two 16B loads.
static __device__ __forceinline__ v16h load_a_frag(const _Float16* row, int k,
                                                   int laneHi8) {
  v16h a;
  uint4* ap = (uint4*)&a;
  ap[0] = *(const uint4*)(row + k + 2 * laneHi8);
  ap[1] = *(const uint4*)(row + k + 2 * laneHi8 + 8);
  return a;
}

// B fragment, this code's contiguous k order (same as A): lane n<16 = column n, K=k..k+15 contiguous;
// lane n+16 = column n, K=k+16..k+31. One 32B contiguous load from the row
// holding column n (row n of the transposed operand).
static __device__ __forceinline__ v16h load_b_frag(const _Float16* colrow,
                                                   int k, int khalf16) {
  return *(const v16h*)(colrow + k + khalf16);
}

// -------------------------------- prep kernels -----------------------------

__global__ void k_f32_to_f16(const float* __restrict__ in,
                             _Float16* __restrict__ out, int n4) {
  int i = blockIdx.x * blockDim.x + threadIdx.x;
  if (i < n4) {
    float4 v = ((const float4*)in)[i];
    _Float16* o = out + (size_t)i * 4;
    o[0] = (_Float16)v.x; o[1] = (_Float16)v.y;
    o[2] = (_Float16)v.z; o[3] = (_Float16)v.w;
  }
}

__global__ void k_mask(const int* __restrict__ usage, float* __restrict__ mf,
                       int n) {
  int i = blockIdx.x * blockDim.x + threadIdx.x;
  if (i < n) mf[i] = (usage[i] > 0) ? 1.0f : 0.0f;
}

// One block (256 threads) per row of length 1024: out = in * rsqrt(sumsq+eps) * scale
__global__ __launch_bounds__(256) void k_norm_rows(const float* __restrict__ in,
                                                   _Float16* __restrict__ out,
                                                   float scale) {
  __shared__ float red[256];
  const int row = blockIdx.x;
  const float* p = in + (size_t)row * HDIM;
  float v[4], ss = 0.0f;
  for (int i = 0; i < 4; ++i) {
    v[i] = p[threadIdx.x + i * 256];
    ss += v[i] * v[i];
  }
  red[threadIdx.x] = ss;
  __syncthreads();
  for (int s = 128; s > 0; s >>= 1) {
    if (threadIdx.x < s) red[threadIdx.x] += red[threadIdx.x + s];
    __syncthreads();
  }
  float rs = rsqrtf(red[0] + 1e-6f) * scale;
  _Float16* o = out + (size_t)row * HDIM;
  for (int i = 0; i < 4; ++i)
    o[threadIdx.x + i * 256] = (_Float16)(v[i] * rs);
}

// V [M,H] f32  ->  Vt [H,M] f16   (LDS 32x32 tile transpose, block 32x8)
__global__ __launch_bounds__(256) void k_transpose_f16(
    const float* __restrict__ V, _Float16* __restrict__ Vt) {
  __shared__ _Float16 t[32][33];
  const int m0 = blockIdx.x * 32, h0 = blockIdx.y * 32;
  for (int i = 0; i < 4; ++i) {
    int r = threadIdx.y + i * 8;
    t[r][threadIdx.x] = (_Float16)V[(size_t)(m0 + r) * HDIM + h0 + threadIdx.x];
  }
  __syncthreads();
  for (int i = 0; i < 4; ++i) {
    int r = threadIdx.y + i * 8;
    Vt[(size_t)(h0 + r) * MSLOTS + m0 + threadIdx.x] = t[threadIdx.x][r];
  }
}

// ------------------------- generic WMMA GEMM: C = A @ W^T + bias -----------
// A [Nrows,1024] f16, W [1024,1024] f16 (row c = output column c), C f32.
// 512 threads = 16 waves; block = 16 rows; wave j covers cols j*64..+63.
// A tile staged in LDS; K-loop double-buffered (load k+32 while computing k).
__global__ __launch_bounds__(512) void k_gemm_nt(
    const _Float16* __restrict__ A, const _Float16* __restrict__ W,
    const float* __restrict__ bias, float* __restrict__ C) {
  __shared__ __align__(32) _Float16 a_s[16 * HDIM];  // 32 KB
  const int tid     = threadIdx.x;
  const int lane    = tid & 31;
  const int wave    = tid >> 5;
  const int lrow    = lane & 15;
  const int laneHi8 = (lane >> 4) * 8;
  const int khalf   = (lane >> 4) * 16;
  const int rowbase = blockIdx.x * 16;

  {  // stage 16x1024 A tile
    const uint4* src = (const uint4*)(A + (size_t)rowbase * HDIM);
    uint4* dst = (uint4*)a_s;
    for (int i = tid; i < 16 * HDIM / 8; i += 512) dst[i] = src[i];
  }
  __syncthreads();

  const v8f zero = {0, 0, 0, 0, 0, 0, 0, 0};
  v8f acc[4];
#pragma unroll
  for (int t = 0; t < 4; ++t) acc[t] = zero;

  const _Float16* Arow  = a_s + (size_t)lrow * HDIM;
  const _Float16* Wbase = W + (size_t)(wave * 64 + lrow) * HDIM;

  v16h a[2], b[2][4];
  a[0] = load_a_frag(Arow, 0, laneHi8);
#pragma unroll
  for (int t = 0; t < 4; ++t)
    b[0][t] = load_b_frag(Wbase + (size_t)t * 16 * HDIM, 0, khalf);

  for (int k = 0; k < HDIM; k += 64) {
    a[1] = load_a_frag(Arow, k + 32, laneHi8);
#pragma unroll
    for (int t = 0; t < 4; ++t)
      b[1][t] = load_b_frag(Wbase + (size_t)t * 16 * HDIM, k + 32, khalf);
#pragma unroll
    for (int t = 0; t < 4; ++t) acc[t] = wmma_f16(a[0], b[0][t], acc[t]);
    if (k + 64 < HDIM) {
      a[0] = load_a_frag(Arow, k + 64, laneHi8);
#pragma unroll
      for (int t = 0; t < 4; ++t)
        b[0][t] = load_b_frag(Wbase + (size_t)t * 16 * HDIM, k + 64, khalf);
    }
#pragma unroll
    for (int t = 0; t < 4; ++t) acc[t] = wmma_f16(a[1], b[1][t], acc[t]);
  }

  const int mof = (lane >> 4) * 8;  // C layout: VGPR r = row r (+8 hi lanes)
  for (int t = 0; t < 4; ++t) {
    int col = wave * 64 + t * 16 + lrow;
    float bv = bias[col];
#pragma unroll
    for (int r = 0; r < 8; ++r) {
      int row = rowbase + r + mof;
      C[(size_t)row * HDIM + col] = acc[t][r] + bv;
    }
  }
}

// --------------------- fused flash attention over memory -------------------
// qn  [8192,1024] f16 (row-normalized, pre-scaled by 1/T)
// mkn [16384,1024] f16 (row-normalized memory keys)
// Vt  [1024,16384] f16 (transposed memory values)
// maskf [16384] f32 in {0,1};  retr [8192,1024] f16 output
// 512 threads (16 waves), 32 query rows per block, M in 256-slot chunks.
// Wave j: scores for slots chunk+j*16..+15 (phase 1); retrieved cols
// j*64..+63 (phase 3). Both WMMA loops are double-buffered.
__global__ __launch_bounds__(512) void k_flash(
    const _Float16* __restrict__ qn, const _Float16* __restrict__ mkn,
    const _Float16* __restrict__ Vt, const float* __restrict__ maskf,
    _Float16* __restrict__ retr) {
  constexpr int CH = 256;
  __shared__ __align__(32) _Float16 qn_s[32 * HDIM];  // 64 KB
  __shared__ __align__(16) float    s_s[32 * CH];     // 32 KB raw scores
  __shared__ __align__(32) _Float16 w_s[32 * CH];     // 16 KB exp weights
  __shared__ float runmax[32], runsum[32], fact[32];

  const int tid     = threadIdx.x;
  const int lane    = tid & 31;
  const int wave    = tid >> 5;
  const int lrow    = lane & 15;
  const int laneHi8 = (lane >> 4) * 8;
  const int khalf   = (lane >> 4) * 16;
  const int mof     = (lane >> 4) * 8;
  const int qbase   = blockIdx.x * 32;

  {  // stage 32x1024 f16 qn tile into LDS
    const uint4* src = (const uint4*)(qn + (size_t)qbase * HDIM);
    uint4* dst = (uint4*)qn_s;
    for (int i = tid; i < 32 * HDIM / 8; i += 512) dst[i] = src[i];
  }
  if (tid < 32) { runmax[tid] = -3.0e38f; runsum[tid] = 0.0f; }
  __syncthreads();

  const v8f zero = {0, 0, 0, 0, 0, 0, 0, 0};
  v8f acc_r[2][4];
#pragma unroll
  for (int rg = 0; rg < 2; ++rg)
#pragma unroll
    for (int t = 0; t < 4; ++t) acc_r[rg][t] = zero;

  const _Float16* q0 = qn_s + (size_t)lrow * HDIM;
  const _Float16* q1 = qn_s + (size_t)(16 + lrow) * HDIM;
  const _Float16* w0 = w_s + (size_t)lrow * CH;
  const _Float16* w1 = w_s + (size_t)(16 + lrow) * CH;

  for (int chunk = 0; chunk < MSLOTS; chunk += CH) {
    // ---- phase 1: scores. Wave j owns slots chunk + j*16 .. +15, K = 1024.
    const int slot = chunk + wave * 16 + lrow;
    const _Float16* Brow = mkn + (size_t)slot * HDIM;
    if (chunk + CH < MSLOTS) {  // L2 prefetch hints for next chunk
      __builtin_prefetch(mkn + (size_t)(slot + CH) * HDIM, 0, 0);
      __builtin_prefetch(Vt + (size_t)(wave * 64 + lrow) * MSLOTS + chunk + CH, 0, 0);
    }
    v8f sc0 = zero, sc1 = zero;
    {
      v16h sb[2], sa0[2], sa1[2];
      sb[0]  = load_b_frag(Brow, 0, khalf);
      sa0[0] = load_a_frag(q0, 0, laneHi8);
      sa1[0] = load_a_frag(q1, 0, laneHi8);
      for (int k = 0; k < HDIM; k += 64) {
        sb[1]  = load_b_frag(Brow, k + 32, khalf);
        sa0[1] = load_a_frag(q0, k + 32, laneHi8);
        sa1[1] = load_a_frag(q1, k + 32, laneHi8);
        sc0 = wmma_f16(sa0[0], sb[0], sc0);
        sc1 = wmma_f16(sa1[0], sb[0], sc1);
        if (k + 64 < HDIM) {
          sb[0]  = load_b_frag(Brow, k + 64, khalf);
          sa0[0] = load_a_frag(q0, k + 64, laneHi8);
          sa1[0] = load_a_frag(q1, k + 64, laneHi8);
        }
        sc0 = wmma_f16(sa0[1], sb[1], sc0);
        sc1 = wmma_f16(sa1[1], sb[1], sc1);
      }
    }
    // mask (exactly NEG_INF for unused slots, matching reference) + spill to LDS
    const float msk = maskf[slot];
    const int sl = wave * 16 + lrow;
#pragma unroll
    for (int r = 0; r < 8; ++r) {
      s_s[(r + mof) * CH + sl]      = (msk > 0.0f) ? sc0[r] : NEG_INF;
      s_s[(16 + r + mof) * CH + sl] = (msk > 0.0f) ? sc1[r] : NEG_INF;
    }
    __syncthreads();

    // ---- phase 2: cooperative online softmax (32 rows x 256 slots).
    {
      const int row = tid >> 4, seg = tid & 15;   // 16 threads/row, 16 slots each
      float* sp = s_s + row * CH + seg * 16;
      float v[16], pm = -3.0e38f;
      for (int i = 0; i < 16; ++i) { v[i] = sp[i]; pm = fmaxf(pm, v[i]); }
      for (int off = 1; off < 16; off <<= 1) pm = fmaxf(pm, __shfl_xor(pm, off, 16));
      const float nm = fmaxf(runmax[row], pm);
      float ps = 0.0f;
      _Float16* wp = w_s + row * CH + seg * 16;
      for (int i = 0; i < 16; ++i) {
        float e = __expf(v[i] - nm);
        ps += e;
        wp[i] = (_Float16)e;
      }
      for (int off = 1; off < 16; off <<= 1) ps += __shfl_xor(ps, off, 16);
      if (seg == 0) {
        float f = __expf(runmax[row] - nm);
        fact[row]   = f;
        runsum[row] = runsum[row] * f + ps;
        runmax[row] = nm;
      }
    }
    __syncthreads();

    // ---- phase 3: rescale accumulators, then retrieved += w @ V chunk.
#pragma unroll
    for (int rg = 0; rg < 2; ++rg)
#pragma unroll
      for (int r = 0; r < 8; ++r) {
        float fm = fact[rg * 16 + mof + r];
#pragma unroll
        for (int t = 0; t < 4; ++t) acc_r[rg][t][r] *= fm;
      }
    {
      const _Float16* Vbase = Vt + (size_t)(wave * 64 + lrow) * MSLOTS + chunk;
      v16h pa0[2], pa1[2], pb[2][4];
      pa0[0] = load_a_frag(w0, 0, laneHi8);
      pa1[0] = load_a_frag(w1, 0, laneHi8);
#pragma unroll
      for (int t = 0; t < 4; ++t)
        pb[0][t] = load_b_frag(Vbase + (size_t)t * 16 * MSLOTS, 0, khalf);
      for (int ks = 0; ks < CH; ks += 64) {
        pa0[1] = load_a_frag(w0, ks + 32, laneHi8);
        pa1[1] = load_a_frag(w1, ks + 32, laneHi8);
#pragma unroll
        for (int t = 0; t < 4; ++t)
          pb[1][t] = load_b_frag(Vbase + (size_t)t * 16 * MSLOTS, ks + 32, khalf);
#pragma unroll
        for (int t = 0; t < 4; ++t) {
          acc_r[0][t] = wmma_f16(pa0[0], pb[0][t], acc_r[0][t]);
          acc_r[1][t] = wmma_f16(pa1[0], pb[0][t], acc_r[1][t]);
        }
        if (ks + 64 < CH) {
          pa0[0] = load_a_frag(w0, ks + 64, laneHi8);
          pa1[0] = load_a_frag(w1, ks + 64, laneHi8);
#pragma unroll
          for (int t = 0; t < 4; ++t)
            pb[0][t] = load_b_frag(Vbase + (size_t)t * 16 * MSLOTS, ks + 64, khalf);
        }
#pragma unroll
        for (int t = 0; t < 4; ++t) {
          acc_r[0][t] = wmma_f16(pa0[1], pb[1][t], acc_r[0][t]);
          acc_r[1][t] = wmma_f16(pa1[1], pb[1][t], acc_r[1][t]);
        }
      }
    }
    __syncthreads();  // protect s_s/w_s/fact before next chunk
  }

  // ---- finalize: retrieved = acc / runsum, store f16 for the output GEMM.
  for (int rg = 0; rg < 2; ++rg)
    for (int r = 0; r < 8; ++r) {
      int m = rg * 16 + mof + r;
      float inv = 1.0f / runsum[m];
      for (int t = 0; t < 4; ++t) {
        int col = wave * 64 + t * 16 + lrow;
        retr[(size_t)(qbase + m) * HDIM + col] = (_Float16)(acc_r[rg][t][r] * inv);
      }
    }
}

// --------------------------------- launch ----------------------------------

extern "C" void kernel_launch(void* const* d_in, const int* in_sizes, int n_in,
                              void* d_out, int out_size, void* d_ws,
                              size_t ws_size, hipStream_t stream) {
  const float* hs  = (const float*)d_in[0];   // [8,1024,1024]
  const float* Wk  = (const float*)d_in[1];   // [1024,1024]
  const float* bk  = (const float*)d_in[2];   // [1024]
  // d_in[3]=Wv, d_in[4]=bv are unused by the reference.
  const float* Wo  = (const float*)d_in[5];   // [1024,1024]
  const float* bo  = (const float*)d_in[6];   // [1024]
  const float* mk  = (const float*)d_in[7];   // [16384,1024]
  const float* mv  = (const float*)d_in[8];   // [16384,1024]
  const int*   usage = (const int*)d_in[9];   // [16384]
  float* out = (float*)d_out;

  // workspace carve-up (~156 MB total)
  char* ws = (char*)d_ws;
  size_t off = 0;
  auto carve = [&](size_t bytes) {
    void* p = ws + off;
    off += (bytes + 255) & ~(size_t)255;
    return p;
  };
  _Float16* hs_h   = (_Float16*)carve((size_t)NROWS * HDIM * 2);
  _Float16* Wk_h   = (_Float16*)carve((size_t)HDIM * HDIM * 2);
  _Float16* Wo_h   = (_Float16*)carve((size_t)HDIM * HDIM * 2);
  _Float16* mkn_h  = (_Float16*)carve((size_t)MSLOTS * HDIM * 2);
  _Float16* Vt_h   = (_Float16*)carve((size_t)MSLOTS * HDIM * 2);
  float*    maskf  = (float*)carve((size_t)MSLOTS * 4);
  float*    q_f    = (float*)carve((size_t)NROWS * HDIM * 4);
  _Float16* qn_h   = (_Float16*)carve((size_t)NROWS * HDIM * 2);
  _Float16* retr_h = (_Float16*)carve((size_t)NROWS * HDIM * 2);

  // 1) f32 -> f16 conversions
  k_f32_to_f16<<<(NROWS * HDIM / 4 + 255) / 256, 256, 0, stream>>>(hs, hs_h,
                                                                   NROWS * HDIM / 4);
  k_f32_to_f16<<<(HDIM * HDIM / 4 + 255) / 256, 256, 0, stream>>>(Wk, Wk_h,
                                                                  HDIM * HDIM / 4);
  k_f32_to_f16<<<(HDIM * HDIM / 4 + 255) / 256, 256, 0, stream>>>(Wo, Wo_h,
                                                                  HDIM * HDIM / 4);
  // 2) usage mask, normalized memory keys, transposed values
  k_mask<<<MSLOTS / 256, 256, 0, stream>>>(usage, maskf, MSLOTS);
  k_norm_rows<<<MSLOTS, 256, 0, stream>>>(mk, mkn_h, 1.0f);
  k_transpose_f16<<<dim3(MSLOTS / 32, HDIM / 32), dim3(32, 8), 0, stream>>>(mv, Vt_h);
  // 3) q = hs @ Wk^T + bk (WMMA), then row-normalize with 1/T baked in
  k_gemm_nt<<<NROWS / 16, 512, 0, stream>>>(hs_h, Wk_h, bk, q_f);
  k_norm_rows<<<NROWS, 256, 0, stream>>>(q_f, qn_h, INV_TEMP);
  // 4) fused scores + mask + online softmax + retrieved (WMMA both GEMMs)
  k_flash<<<NROWS / 32, 512, 0, stream>>>(qn_h, mkn_h, Vt_h, maskf, retr_h);
  // 5) out = retrieved @ Wo^T + bo (WMMA)
  k_gemm_nt<<<NROWS / 16, 512, 0, stream>>>(retr_h, Wo_h, bo, out);
}
